// Attention_29102698398270
// MI455X (gfx1250) — matrix operand order repaired and checked by reading, not compiled, not run on hardware
//
#include <hip/hip_runtime.h>
#include <hip/hip_bf16.h>
#include <stdint.h>

// Sizes fixed by the reference
#define B_   8
#define H_   8
#define L_   1024
#define DK   512     // D_IN
#define DH   64      // head dim
#define MTOT (B_ * L_)   // 8192 rows in the projection GEMMs

typedef __attribute__((ext_vector_type(16))) __bf16 v16bf;
typedef __attribute__((ext_vector_type(8)))  __bf16 v8bf;
typedef __attribute__((ext_vector_type(8)))  float  v8f;

static __device__ inline v16bf join16(v8bf lo, v8bf hi) {
  v16bf r;
#pragma unroll
  for (int i = 0; i < 8; ++i) { r[i] = lo[i]; r[i + 8] = hi[i]; }
  return r;
}

// ---------------------------------------------------------------------------
// Kernel 1: convert W (fp32, [K][N]) -> bf16 transposed [N][K]
// ---------------------------------------------------------------------------
__global__ __launch_bounds__(256)
void mha_wt_convert(const float* __restrict__ w, __bf16* __restrict__ wt) {
  int idx = blockIdx.x * 256 + threadIdx.x;   // idx = n*512 + k
  int n = idx >> 9;
  int k = idx & 511;
  wt[idx] = (__bf16)w[k * DK + n];
}

// ---------------------------------------------------------------------------
// Kernel 2: projection GEMM  out = seq[MTOT,512] @ W[512,512]  (bf16 WMMA)
// Workgroup: 256 threads = 8 waves, tile 128(M) x 64(N), K chunked by 32.
// mode 0: Q -> qk_out[b,h,l,d] * (1/sqrt(64))
// mode 1: K -> qk_out[b,h,l,d]
// mode 2: V -> v_out [b,h,d,l]  (transposed for P@V B-fragments)
// ---------------------------------------------------------------------------
__global__ __launch_bounds__(256)
void mha_proj_gemm(const float* __restrict__ seq, const __bf16* __restrict__ wt,
                   __bf16* __restrict__ qk_out, __bf16* __restrict__ v_out,
                   int mode) {
  __shared__ __align__(16) __bf16 ldsA[128 * 32];  // 8 KB
  __shared__ __align__(16) __bf16 ldsB[64 * 32];   // 4 KB

  const int tid  = threadIdx.x;
  const int lane = tid & 31;
  const int w    = tid >> 5;              // wave id 0..7 (one 16-row strip each)
  const int m0   = blockIdx.x * 128;
  const int n0   = blockIdx.y * 64;

  // WMMA 16-bit fragment lane geometry (ISA 7.12.2)
  const int mA  = lane & 15;
  const int kbA = (lane < 16) ? 0 : 8;    // A: K = kbA..kbA+7 and kbA+16..kbA+23
  const int nB  = lane & 15;
  const int kbB = (lane < 16) ? 0 : 8;    // B: K = kbB..kbB+7 and kbB+16..kbB+23
  const int hi  = lane >> 4;

  v8f acc[4] = {};

  for (int kk = 0; kk < DK; kk += 32) {
    __syncthreads();   // protect LDS from previous iteration's readers

    // --- stage A tile: 128x32 fp32 -> bf16 in LDS (convert in flight)
    {
      int row = tid >> 1, half = tid & 1;
      const float* src = seq + (size_t)(m0 + row) * DK + kk + half * 16;
      __bf16* dst = &ldsA[row * 32 + half * 16];
#pragma unroll
      for (int i = 0; i < 16; ++i) dst[i] = (__bf16)src[i];
    }

    // --- stage B tile: 64x32 bf16 pure copy via async global->LDS (gfx1250 TDM-lite path)
    {
      int row = tid >> 2, seg = tid & 3;
      const __bf16* src = wt + (size_t)(n0 + row) * DK + kk + seg * 8;
      __bf16* dst = &ldsB[row * 32 + seg * 8];
      unsigned lds_off = (unsigned)(size_t)dst;               // low 32 bits = LDS offset
      unsigned long long ga = (unsigned long long)(size_t)src;
      asm volatile("global_load_async_to_lds_b128 %0, %1, off"
                   :: "v"(lds_off), "v"(ga) : "memory");
    }
    asm volatile("s_wait_asynccnt 0x0" ::: "memory");
    __syncthreads();

    // --- fragments + WMMA
    v8bf alo = *(const v8bf*)&ldsA[(w * 16 + mA) * 32 + kbA];
    v8bf ahi = *(const v8bf*)&ldsA[(w * 16 + mA) * 32 + kbA + 16];
    v16bf afr = join16(alo, ahi);
#pragma unroll
    for (int t = 0; t < 4; ++t) {
      v8bf blo = *(const v8bf*)&ldsB[(t * 16 + nB) * 32 + kbB];
      v8bf bhi = *(const v8bf*)&ldsB[(t * 16 + nB) * 32 + kbB + 16];
      v16bf bfr = join16(blo, bhi);
      acc[t] = __builtin_amdgcn_wmma_f32_16x16x32_bf16(
          false, afr, false, bfr, (short)0, acc[t], false, false);
    }
  }

  // --- epilogue: scatter into head-major bf16 workspace
#pragma unroll
  for (int t = 0; t < 4; ++t) {
    int nn = n0 + t * 16 + nB;     // column in [0,512)
    int h = nn >> 6, d = nn & 63;
#pragma unroll
    for (int r = 0; r < 8; ++r) {
      int mg = m0 + w * 16 + r + 8 * hi;    // global row in [0, 8192)
      int b = mg >> 10, l = mg & 1023;
      float val = acc[t][r];
      if (mode == 0) {
        qk_out[((size_t)(b * H_ + h) * L_ + l) * DH + d] = (__bf16)(val * 0.125f);
      } else if (mode == 1) {
        qk_out[((size_t)(b * H_ + h) * L_ + l) * DH + d] = (__bf16)val;
      } else {
        v_out[((size_t)(b * H_ + h) * DH + d) * L_ + l] = (__bf16)val;
      }
    }
  }
}

// ---------------------------------------------------------------------------
// Kernel 3: causal flash attention. One 16-row q tile per wave.
// qws/kws: [B,H,L,64] bf16 (q pre-scaled by 1/8); vws: [B,H,64,L] bf16.
// out: [B, L, 512] fp32.
// ---------------------------------------------------------------------------
__global__ __launch_bounds__(128)
void mha_attn(const __bf16* __restrict__ qws, const __bf16* __restrict__ kws,
              const __bf16* __restrict__ vws, float* __restrict__ out) {
  __shared__ __align__(16) __bf16 ldsP[4][16 * 32];   // wave-private P tile

  const int lane  = threadIdx.x & 31;
  const int wslot = threadIdx.x >> 5;
  const int wid   = blockIdx.x * 4 + wslot;   // 0..4095
  const int qt    = wid & 63;                 // q tile within sequence
  const int bh    = wid >> 6;                 // (b*H + h)
  const int q0    = qt * 16;

  const __bf16* qp = qws + (size_t)bh * L_ * DH;
  const __bf16* kp = kws + (size_t)bh * L_ * DH;
  const __bf16* vp = vws + (size_t)bh * DH * L_;

  const int mA  = lane & 15;
  const int kbA = (lane < 16) ? 0 : 8;
  const int nB  = lane & 15;
  const int kbB = (lane < 16) ? 0 : 8;
  const int hi  = lane >> 4;

  // Q fragments: D=64 -> two K=32 A-fragments
  v16bf aq[2];
#pragma unroll
  for (int f = 0; f < 2; ++f) {
    const __bf16* base = qp + (size_t)(q0 + mA) * DH + f * 32 + kbA;
    aq[f] = join16(*(const v8bf*)base, *(const v8bf*)(base + 16));
  }

  v8f o[4] = {};
  float mrow[8], lrow[8];
#pragma unroll
  for (int r = 0; r < 8; ++r) { mrow[r] = -3.0e38f; lrow[r] = 0.0f; }

  const int jend = q0 + 16;   // causal: kv <= q0+15
  for (int jb = 0; jb < jend; jb += 32) {
    // ---- S = q @ k^T  (two 16x16 score tiles, K-dim = D = 64)
    v8f s0 = {}, s1 = {};
#pragma unroll
    for (int f = 0; f < 2; ++f) {
      const __bf16* b0 = kp + (size_t)(jb + nB) * DH + f * 32 + kbB;
      v16bf bk0 = join16(*(const v8bf*)b0, *(const v8bf*)(b0 + 16));
      s0 = __builtin_amdgcn_wmma_f32_16x16x32_bf16(false, aq[f], false, bk0,
                                                   (short)0, s0, false, false);
      const __bf16* b1 = kp + (size_t)(jb + 16 + nB) * DH + f * 32 + kbB;
      v16bf bk1 = join16(*(const v8bf*)b1, *(const v8bf*)(b1 + 16));
      s1 = __builtin_amdgcn_wmma_f32_16x16x32_bf16(false, aq[f], false, bk1,
                                                   (short)0, s1, false, false);
    }

    // ---- causal mask + online softmax (row stats per C-layout register row)
    const int gk0 = jb + nB, gk1 = gk0 + 16;
#pragma unroll
    for (int r = 0; r < 8; ++r) {
      int gq = q0 + r + 8 * hi;
      float v0 = (gk0 <= gq) ? s0[r] : -1.0e12f;
      float v1 = (gk1 <= gq) ? s1[r] : -1.0e12f;
      float rm = fmaxf(v0, v1);
#pragma unroll
      for (int msk = 1; msk < 16; msk <<= 1)
        rm = fmaxf(rm, __shfl_xor(rm, msk, 32));
      float newm = fmaxf(mrow[r], rm);
      float p0 = __expf(v0 - newm);
      float p1 = __expf(v1 - newm);
      float rs = p0 + p1;
#pragma unroll
      for (int msk = 1; msk < 16; msk <<= 1)
        rs += __shfl_xor(rs, msk, 32);
      float corr = __expf(mrow[r] - newm);
      mrow[r] = newm;
      lrow[r] = lrow[r] * corr + rs;
#pragma unroll
      for (int t = 0; t < 4; ++t) o[t][r] *= corr;
      // C-layout -> row-major bf16 P tile in LDS (layout transpose bounce)
      int m = r + 8 * hi;
      ldsP[wslot][m * 32 + nB]      = (__bf16)p0;
      ldsP[wslot][m * 32 + 16 + nB] = (__bf16)p1;
    }
    __builtin_amdgcn_wave_barrier();   // same-wave LDS ops are in-order; keep scheduler honest

    // ---- P fragment (A, 16x32) from LDS, V fragments (B) from global, accumulate
    const __bf16* pb = &ldsP[wslot][mA * 32 + kbA];
    v16bf ap = join16(*(const v8bf*)pb, *(const v8bf*)(pb + 16));
#pragma unroll
    for (int t = 0; t < 4; ++t) {
      const __bf16* vb = vp + (size_t)(t * 16 + nB) * L_ + jb + kbB;
      v16bf bv = join16(*(const v8bf*)vb, *(const v8bf*)(vb + 16));
      o[t] = __builtin_amdgcn_wmma_f32_16x16x32_bf16(false, ap, false, bv,
                                                     (short)0, o[t], false, false);
    }
  }

  // ---- epilogue: out[b, l, h*64 + d] = o / l_i
  const int b = bh >> 3, h = bh & 7;
#pragma unroll
  for (int t = 0; t < 4; ++t) {
#pragma unroll
    for (int r = 0; r < 8; ++r) {
      int l = q0 + r + 8 * hi;
      out[((size_t)(b * L_ + l)) * 512 + h * 64 + t * 16 + nB] = o[t][r] / lrow[r];
    }
  }
}

// ---------------------------------------------------------------------------
// Host launcher
// ---------------------------------------------------------------------------
extern "C" void kernel_launch(void* const* d_in, const int* in_sizes, int n_in,
                              void* d_out, int out_size, void* d_ws, size_t ws_size,
                              hipStream_t stream) {
  const float* Q  = (const float*)d_in[0];
  const float* K  = (const float*)d_in[1];
  const float* V  = (const float*)d_in[2];
  const float* WQ = (const float*)d_in[3];
  const float* WK = (const float*)d_in[4];
  const float* WV = (const float*)d_in[5];
  float* out = (float*)d_out;

  char* ws = (char*)d_ws;
  const size_t WT_BYTES  = (size_t)DK * DK * sizeof(__bf16);       // 512 KB each
  const size_t HWS_BYTES = (size_t)B_ * H_ * L_ * DH * sizeof(__bf16); // 8 MB each
  __bf16* wtQ = (__bf16*)(ws);
  __bf16* wtK = (__bf16*)(ws + WT_BYTES);
  __bf16* wtV = (__bf16*)(ws + 2 * WT_BYTES);
  __bf16* qws = (__bf16*)(ws + 3 * WT_BYTES);
  __bf16* kws = (__bf16*)(ws + 3 * WT_BYTES + HWS_BYTES);
  __bf16* vws = (__bf16*)(ws + 3 * WT_BYTES + 2 * HWS_BYTES);

  dim3 cgrid((DK * DK) / 256);
  mha_wt_convert<<<cgrid, 256, 0, stream>>>(WQ, wtQ);
  mha_wt_convert<<<cgrid, 256, 0, stream>>>(WK, wtK);
  mha_wt_convert<<<cgrid, 256, 0, stream>>>(WV, wtV);

  dim3 pgrid(MTOT / 128, DK / 64);   // 64 x 8 blocks
  mha_proj_gemm<<<pgrid, 256, 0, stream>>>(Q, wtQ, qws, nullptr, 0);
  mha_proj_gemm<<<pgrid, 256, 0, stream>>>(K, wtK, kws, nullptr, 1);
  mha_proj_gemm<<<pgrid, 256, 0, stream>>>(V, wtV, nullptr, vws, 2);

  // B*H*(L/16) = 4096 q-tiles, 4 waves per block
  mha_attn<<<dim3(4096 / 4), 128, 0, stream>>>(qws, kws, vws, out);
}
